// Berty_21680994910784
// MI455X (gfx1250) — matrix operand order repaired and checked by reading, not compiled, not run on hardware
//
#include <hip/hip_runtime.h>
#include <hip/hip_bf16.h>

// Problem constants (match reference)
#define Bb 16
#define Tt 512
#define Cc 1024
#define Hh 16
#define Dd 64
#define Ll 12
#define Ff 4096

#define USE_ASYNC_LDS 1

typedef __bf16 bf16;
typedef __attribute__((ext_vector_type(16))) __bf16 v16bf;
typedef __attribute__((ext_vector_type(8)))  __bf16 v8bf;
typedef __attribute__((ext_vector_type(2)))  __bf16 v2bf;
typedef __attribute__((ext_vector_type(8)))  float  v8f;

static __device__ __forceinline__ v16bf cat8(v8bf lo, v8bf hi) {
  return __builtin_shufflevector(lo, hi, 0,1,2,3,4,5,6,7,8,9,10,11,12,13,14,15);
}
static __device__ __forceinline__ v8f wmma_bf16(v16bf a, v16bf b, v8f c) {
  // D = A(16x32 bf16) * B(32x16 bf16) + C(16x16 f32)
  return __builtin_amdgcn_wmma_f32_16x16x32_bf16(false, a, false, b, (short)0, c,
                                                 false, false);
}
static __device__ __forceinline__ bf16 f2bf(float x) { return (bf16)x; }

// One b128 global->LDS copy, async (ASYNCcnt) on CDNA5.
static __device__ __forceinline__ void async_copy16(bf16* dstLds, const bf16* src) {
#if USE_ASYNC_LDS
  unsigned off = (unsigned)(size_t)dstLds;  // LDS aperture: addr[31:0] = LDS offset
  asm volatile("global_load_async_to_lds_b128 %0, %1, off"
               :: "v"(off), "v"(src) : "memory");
#else
  *(v8bf*)dstLds = *(const v8bf*)src;
#endif
}

// Stage one 32(K) x 256(N) fp32 W tile as bf16, transposed, K-pairs packed
// so each WMMA B-fragment is a single contiguous 32B LDS read.
static __device__ __forceinline__ void stage_w_tile(
    bf16 (*BsT)[32], const float* w0row, int kp, int c0, int N) {
  const float4* s0 = (const float4*)w0row;
  const float4* s1 = (const float4*)(w0row + N);
  float4 a0 = s0[0], a1 = s0[1], a2 = s0[2], a3 = s0[3];
  float4 b0 = s1[0], b1 = s1[1], b2 = s1[2], b3 = s1[3];
  float r0[16] = {a0.x,a0.y,a0.z,a0.w, a1.x,a1.y,a1.z,a1.w,
                  a2.x,a2.y,a2.z,a2.w, a3.x,a3.y,a3.z,a3.w};
  float r1[16] = {b0.x,b0.y,b0.z,b0.w, b1.x,b1.y,b1.z,b1.w,
                  b2.x,b2.y,b2.z,b2.w, b3.x,b3.y,b3.z,b3.w};
#pragma unroll
  for (int i = 0; i < 16; ++i) {
    v2bf pr; pr[0] = f2bf(r0[i]); pr[1] = f2bf(r1[i]);
    *(v2bf*)&BsT[c0 + i][kp * 2] = pr;  // merged into ds_store_2addr_b32
  }
}

// ---------------------------------------------------------------------------
// Embedding: x[b*T+t, c] = tok_emb[idx[b,t], c] + pos_emb[t, c]
// ---------------------------------------------------------------------------
__global__ __launch_bounds__(256) void embed_kernel(
    const int* __restrict__ idx, const float* __restrict__ tok,
    const float* __restrict__ pos, float* __restrict__ x) {
  int row = blockIdx.x;          // 0..B*T-1
  int t   = row % Tt;
  int token = idx[row];
  const float* tr = tok + (size_t)token * Cc;
  const float* pr = pos + (size_t)t * Cc;
  float* xr = x + (size_t)row * Cc;
  for (int c = threadIdx.x; c < Cc; c += 256) xr[c] = tr[c] + pr[c];
}

// ---------------------------------------------------------------------------
// LayerNorm over C, fp32 in, bf16 out (feeds WMMA GEMMs)
// ---------------------------------------------------------------------------
__global__ __launch_bounds__(256) void layernorm_kernel(
    const float* __restrict__ x, const float* __restrict__ w,
    const float* __restrict__ b, bf16* __restrict__ out) {
  __shared__ float red[256];
  int row = blockIdx.x;
  const float* xr = x + (size_t)row * Cc;
  int tid = threadIdx.x;

  float s = 0.f;
  for (int c = tid; c < Cc; c += 256) s += xr[c];
  red[tid] = s; __syncthreads();
  for (int off = 128; off > 0; off >>= 1) {
    if (tid < off) red[tid] += red[tid + off];
    __syncthreads();
  }
  float mean = red[0] * (1.0f / Cc);
  __syncthreads();

  float v = 0.f;
  for (int c = tid; c < Cc; c += 256) { float d = xr[c] - mean; v += d * d; }
  red[tid] = v; __syncthreads();
  for (int off = 128; off > 0; off >>= 1) {
    if (tid < off) red[tid] += red[tid + off];
    __syncthreads();
  }
  float rstd = rsqrtf(red[0] * (1.0f / Cc) + 1e-5f);

  bf16* orow = out + (size_t)row * Cc;
  for (int c = tid; c < Cc; c += 256)
    orow[c] = f2bf((xr[c] - mean) * rstd * w[c] + b[c]);
}

// ---------------------------------------------------------------------------
// Fused GEMM: out = act(A_bf16[M,K] @ W_f32[K,N] + bias) (+ resid)
//   * 256 threads = 8 waves (2x4), block tile 64(M) x 256(N), K-step 32
//   * Wave tile 32x64: 2 A-frags x 4 B-frags = 8 back-to-back WMMAs / K-step
//   * DOUBLE-BUFFERED LDS: next A tile via GLOBAL_LOAD_ASYNC_TO_LDS_B128
//     overlaps with current-tile WMMAs; one barrier per K-step
//   * Running-pointer addressing; unconditional speculative prefetch of the
//     next-next W rows (global_prefetch_b8)
//   * act: 0 = none, 1 = exact GELU
// ---------------------------------------------------------------------------
__global__ __launch_bounds__(256) void gemm_kernel(
    const bf16*  __restrict__ A,     // [M,K] bf16 row-major
    const float* __restrict__ W,     // [K,N] f32 row-major
    const float* __restrict__ bias,  // [N]
    const float* __restrict__ resid, // [M,N] or null
    float* __restrict__ outF,        // [M,N] or null
    bf16*  __restrict__ outB,        // [M,N] or null
    int N, int K, int act) {
  __shared__ __align__(32) bf16 As[2][64][32];    // 8 KB
  __shared__ __align__(64) bf16 BsT[2][256][32];  // 32 KB, [col][k] (K-pairs)

  int tid  = threadIdx.x;
  int wave = tid >> 5, lane = tid & 31;
  int mw = wave & 1, nw = wave >> 1;   // 2 x 4 wave grid
  int hl = lane >> 4, ln = lane & 15;  // half / lane-in-half

  long rowBase = (long)blockIdx.y * 64;
  long colBase = (long)blockIdx.x * 256;

  v8f acc[2][4] = {};

  // staging assignments
  int ar = tid >> 2;            // 0..63   A row
  int ac = (tid & 3) * 8;       // 0,8,16,24
  int kp = tid >> 4;            // 0..15   W k-pair (rows 2kp, 2kp+1)
  int c0 = (tid & 15) * 16;     // 0..240  W column group

  const bf16*  aCur = A + (rowBase + ar) * (long)K + ac;      // +=32 per step
  const float* wCur = W + (long)(kp * 2) * N + colBase + c0;  // +=32*N per step
  const long   wStep = 32L * N;

  // -------- prologue: stage buffer 0 --------
  async_copy16(&As[0][ar][ac], aCur);
  stage_w_tile(BsT[0], wCur, kp, c0, N);
#if USE_ASYNC_LDS
  __builtin_amdgcn_s_wait_asynccnt(0);
#endif
  __syncthreads();

  // One pipelined step: compute from buffer CUR, stage buffer CUR^1.
  auto step = [&](int CUR, int k0) {
    bool last = (k0 + 32 >= K);
    const bf16*  aNext = last ? aCur : (aCur + 32);          // branchless clamp
    const float* wNext = last ? wCur : (wCur + wStep);

    // issue next A tile async copy first: overlaps with WMMAs below
    async_copy16(&As[CUR ^ 1][ar][ac], aNext);
    // speculative prefetch of next-next W rows (overshoot is dropped by HW)
    __builtin_prefetch(wNext + wStep, 0, 1);

    // load all fragments, then issue 8 WMMAs back-to-back
    v8bf a0lo = *(const v8bf*)&As[CUR][mw * 32 + ln][hl * 16];
    v8bf a0hi = *(const v8bf*)&As[CUR][mw * 32 + ln][hl * 16 + 8];
    v8bf a1lo = *(const v8bf*)&As[CUR][mw * 32 + 16 + ln][hl * 16];
    v8bf a1hi = *(const v8bf*)&As[CUR][mw * 32 + 16 + ln][hl * 16 + 8];
    v16bf af0 = cat8(a0lo, a0hi), af1 = cat8(a1lo, a1hi);
    v16bf bf0 = *(const v16bf*)&BsT[CUR][nw * 64 +  0 + ln][hl * 16];
    v16bf bf1 = *(const v16bf*)&BsT[CUR][nw * 64 + 16 + ln][hl * 16];
    v16bf bf2 = *(const v16bf*)&BsT[CUR][nw * 64 + 32 + ln][hl * 16];
    v16bf bf3 = *(const v16bf*)&BsT[CUR][nw * 64 + 48 + ln][hl * 16];

    acc[0][0] = wmma_bf16(af0, bf0, acc[0][0]);
    acc[1][0] = wmma_bf16(af1, bf0, acc[1][0]);
    acc[0][1] = wmma_bf16(af0, bf1, acc[0][1]);
    acc[1][1] = wmma_bf16(af1, bf1, acc[1][1]);
    acc[0][2] = wmma_bf16(af0, bf2, acc[0][2]);
    acc[1][2] = wmma_bf16(af1, bf2, acc[1][2]);
    acc[0][3] = wmma_bf16(af0, bf3, acc[0][3]);
    acc[1][3] = wmma_bf16(af1, bf3, acc[1][3]);

    // stage next W tile while WMMAs drain
    stage_w_tile(BsT[CUR ^ 1], wNext, kp, c0, N);
#if USE_ASYNC_LDS
    __builtin_amdgcn_s_wait_asynccnt(0);
#endif
    __syncthreads();
    aCur = aNext; wCur = wNext;
  };

  // K is a multiple of 64 for all uses (1024, 4096): two steps per trip keeps
  // the buffer index a compile-time constant.
  for (int k0 = 0; k0 < K; k0 += 64) {
    step(0, k0);
    step(1, k0 + 32);
  }

  // --- epilogue: bias (+gelu) (+residual), f32 and/or bf16 out ---
#pragma unroll
  for (int m = 0; m < 2; ++m) {
#pragma unroll
    for (int t = 0; t < 4; ++t) {
#pragma unroll
      for (int r = 0; r < 8; ++r) {
        long row = rowBase + mw * 32 + m * 16 + r + hl * 8;
        long col = colBase + nw * 64 + t * 16 + ln;
        float val = acc[m][t][r] + bias[col];
        if (act == 1) val = 0.5f * val * (1.0f + erff(val * 0.70710678118654752f));
        if (resid) val += resid[row * N + col];
        if (outF) outF[row * N + col] = val;
        if (outB) outB[row * N + col] = f2bf(val);
      }
    }
  }
}

// ---------------------------------------------------------------------------
// Attention: one block (4 waves, 128 thr) per (b, h, 16-query tile).
//   S = (q k^T)/8 via WMMA -> LDS f32, softmax, P->bf16, Y = P @ V via WMMA.
// q/k/v are [B*T, C] bf16 with head h occupying cols [h*64, h*64+64).
// ---------------------------------------------------------------------------
__global__ __launch_bounds__(128) void attention_kernel(
    const bf16* __restrict__ q, const bf16* __restrict__ k,
    const bf16* __restrict__ v, bf16* __restrict__ y) {
  __shared__ float S[16][Tt];      // 32 KB scores
  __shared__ bf16  P[16][Tt];      // 16 KB probabilities (bf16)
  __shared__ float redm[16][8], reds[16][8], rowstat[16][2];

  int tid  = threadIdx.x;
  int wave = tid >> 5, lane = tid & 31;
  int hl = lane >> 4, ln = lane & 15;

  int blk = blockIdx.x;            // 0 .. B*H*(T/16)-1
  int qt = blk % (Tt / 16);
  int bh = blk / (Tt / 16);
  int h = bh % Hh, b = bh / Hh;
  long base = (long)b * Tt;        // token-row base for this batch
  int q0 = qt * 16;

  const bf16* qp = q + (base + q0) * (long)Cc + h * Dd;

  // ---- S = q k^T * 1/sqrt(D); wave w covers columns [w*128, w*128+128) ----
  for (int jt = 0; jt < 8; ++jt) {
    int j0 = wave * 128 + jt * 16;
    v8f acc = {};
#pragma unroll
    for (int ds = 0; ds < 2; ++ds) {   // D=64 -> two K=32 steps
      v8bf alo = *(const v8bf*)(qp + (long)ln * Cc + ds * 32 + hl * 16);
      v8bf ahi = *(const v8bf*)(qp + (long)ln * Cc + ds * 32 + hl * 16 + 8);
      v16bf af = cat8(alo, ahi);
      // B = k^T: lane = key token column, K contiguous along d
      v16bf bfr = *(const v16bf*)(k + (base + j0 + ln) * (long)Cc + h * Dd +
                                  ds * 32 + hl * 16);
      acc = wmma_bf16(af, bfr, acc);
    }
#pragma unroll
    for (int r = 0; r < 8; ++r) S[r + hl * 8][j0 + ln] = acc[r] * 0.125f;
  }
  __syncthreads();

  // ---- softmax over 512 keys (8 segments of 64 per row) ----
  {
    int row = tid & 15, seg = tid >> 4;
    float m = -1e30f;
    for (int c = seg * 64; c < seg * 64 + 64; ++c) m = fmaxf(m, S[row][c]);
    redm[row][seg] = m; __syncthreads();
    if (tid < 16) {
      float mm = redm[tid][0];
      for (int i = 1; i < 8; ++i) mm = fmaxf(mm, redm[tid][i]);
      rowstat[tid][0] = mm;
    }
    __syncthreads();
    float mm = rowstat[row][0];
    float ssum = 0.f;
    for (int c = seg * 64; c < seg * 64 + 64; ++c) {
      float e = expf(S[row][c] - mm);
      S[row][c] = e; ssum += e;
    }
    reds[row][seg] = ssum; __syncthreads();
    if (tid < 16) {
      float t2 = 0.f;
      for (int i = 0; i < 8; ++i) t2 += reds[tid][i];
      rowstat[tid][1] = 1.0f / t2;
    }
    __syncthreads();
    float rinv = rowstat[row][1];
    for (int c = seg * 64; c < seg * 64 + 64; ++c)
      P[row][c] = f2bf(S[row][c] * rinv);
  }
  __syncthreads();

  // ---- Y = P @ V; wave w computes head-dim columns [w*16, w*16+16) ----
  {
    v8f acc = {};
    for (int kt = 0; kt < 16; ++kt) {  // 512 / 32
      v8bf alo = *(const v8bf*)&P[ln][kt * 32 + hl * 16];
      v8bf ahi = *(const v8bf*)&P[ln][kt * 32 + hl * 16 + 8];
      v16bf af = cat8(alo, ahi);
      v16bf bfr;
#pragma unroll
      for (int e = 0; e < 16; ++e)
        bfr[e] = v[(base + kt * 32 + hl * 16 + e) * (long)Cc + h * Dd +
                   wave * 16 + ln];
      acc = wmma_bf16(af, bfr, acc);
    }
#pragma unroll
    for (int r = 0; r < 8; ++r)
      y[(base + q0 + r + hl * 8) * (long)Cc + h * Dd + wave * 16 + ln] =
          f2bf(acc[r]);
  }
}

// ---------------------------------------------------------------------------
// Head: out[row] = softplus( dot(h[row,:], head_w) + head_b )
// ---------------------------------------------------------------------------
__global__ __launch_bounds__(256) void head_kernel(
    const bf16* __restrict__ hx, const float* __restrict__ hw,
    const float* __restrict__ hb, float* __restrict__ out) {
  __shared__ float red[256];
  int row = blockIdx.x, tid = threadIdx.x;
  const bf16* hr = hx + (size_t)row * Cc;
  float s = 0.f;
  for (int c = tid; c < Cc; c += 256) s += (float)hr[c] * hw[c];
  red[tid] = s; __syncthreads();
  for (int off = 128; off > 0; off >>= 1) {
    if (tid < off) red[tid] += red[tid + off];
    __syncthreads();
  }
  if (tid == 0) {
    float z = red[0] + hb[0];
    out[row] = (z > 20.0f) ? z : log1pf(expf(z));  // softplus
  }
}

// ---------------------------------------------------------------------------
extern "C" void kernel_launch(void* const* d_in, const int* in_sizes, int n_in,
                              void* d_out, int out_size, void* d_ws,
                              size_t ws_size, hipStream_t stream) {
  (void)in_sizes; (void)n_in; (void)out_size; (void)ws_size;
  const int*   idx   = (const int*)  d_in[0];
  const float* tok   = (const float*)d_in[1];
  const float* pos   = (const float*)d_in[2];
  const float* ln1w  = (const float*)d_in[3];
  const float* ln1b  = (const float*)d_in[4];
  const float* ln2w  = (const float*)d_in[5];
  const float* ln2b  = (const float*)d_in[6];
  const float* Wq    = (const float*)d_in[7];
  const float* bq    = (const float*)d_in[8];
  const float* Wk    = (const float*)d_in[9];
  const float* bk    = (const float*)d_in[10];
  const float* Wv    = (const float*)d_in[11];
  const float* bv    = (const float*)d_in[12];
  const float* Wo    = (const float*)d_in[13];
  const float* bo    = (const float*)d_in[14];
  const float* W1    = (const float*)d_in[15];
  const float* b1    = (const float*)d_in[16];
  const float* W2    = (const float*)d_in[17];
  const float* b2    = (const float*)d_in[18];
  const float* lnfw  = (const float*)d_in[19];
  const float* lnfb  = (const float*)d_in[20];
  const float* headw = (const float*)d_in[21];
  const float* headb = (const float*)d_in[22];
  float* out = (float*)d_out;

  const size_t MT = (size_t)Bb * Tt;  // 8192 token rows
  char* p = (char*)d_ws;
  float* x  = (float*)p; p += MT * Cc * sizeof(float);  // fp32 residual stream
  bf16*  hb = (bf16*)p;  p += MT * Cc * sizeof(bf16);   // LN output
  bf16*  qb = (bf16*)p;  p += MT * Cc * sizeof(bf16);
  bf16*  kb = (bf16*)p;  p += MT * Cc * sizeof(bf16);
  bf16*  vb = (bf16*)p;  p += MT * Cc * sizeof(bf16);
  bf16*  yb = (bf16*)p;  p += MT * Cc * sizeof(bf16);   // attention output
  bf16*  gb = (bf16*)p;  p += MT * Ff * sizeof(bf16);   // MLP hidden

  embed_kernel<<<dim3((unsigned)MT), 256, 0, stream>>>(idx, tok, pos, x);

  dim3 gC(Cc / 256, (unsigned)(MT / 64));  // N = C GEMMs (4 x 128)
  dim3 gF(Ff / 256, (unsigned)(MT / 64));  // N = F GEMM (16 x 128)
  dim3 gLN((unsigned)MT);
  dim3 gAtt(Bb * Hh * (Tt / 16));

  for (int l = 0; l < Ll; ++l) {
    const float* wq = Wq + (size_t)l * Cc * Cc;
    const float* wk = Wk + (size_t)l * Cc * Cc;
    const float* wv = Wv + (size_t)l * Cc * Cc;
    const float* wo = Wo + (size_t)l * Cc * Cc;
    const float* w1 = W1 + (size_t)l * Cc * Ff;
    const float* w2 = W2 + (size_t)l * Ff * Cc;

    layernorm_kernel<<<gLN, 256, 0, stream>>>(x, ln1w + l * Cc, ln1b + l * Cc, hb);
    gemm_kernel<<<gC, 256, 0, stream>>>(hb, wq, bq + l * Cc, nullptr, nullptr, qb, Cc, Cc, 0);
    gemm_kernel<<<gC, 256, 0, stream>>>(hb, wk, bk + l * Cc, nullptr, nullptr, kb, Cc, Cc, 0);
    gemm_kernel<<<gC, 256, 0, stream>>>(hb, wv, bv + l * Cc, nullptr, nullptr, vb, Cc, Cc, 0);
    attention_kernel<<<gAtt, 128, 0, stream>>>(qb, kb, vb, yb);
    gemm_kernel<<<gC, 256, 0, stream>>>(yb, wo, bo + l * Cc, x, x, nullptr, Cc, Cc, 0);
    layernorm_kernel<<<gLN, 256, 0, stream>>>(x, ln2w + l * Cc, ln2b + l * Cc, hb);
    gemm_kernel<<<gF, 256, 0, stream>>>(hb, w1, b1 + l * Ff, nullptr, nullptr, gb, Ff, Cc, 1);
    gemm_kernel<<<gC, 256, 0, stream>>>(gb, w2, b2 + l * Cc, x, x, nullptr, Cc, Ff, 0);
  }

  layernorm_kernel<<<gLN, 256, 0, stream>>>(x, lnfw, lnfb, hb);
  head_kernel<<<gLN, 256, 0, stream>>>(hb, headw, headb, out);
}
